// PointNextEncoder_11407433138472
// MI455X (gfx1250) — matrix operand order repaired and checked by reading, not compiled, not run on hardware
//
#include <hip/hip_runtime.h>
#include <math.h>

typedef __attribute__((ext_vector_type(16))) _Float16 v16h;
typedef __attribute__((ext_vector_type(8)))  _Float16 v8h;
typedef __attribute__((ext_vector_type(8)))  float    v8f;

#define NEG_ENC 0x007FFFFFu  // order-encoding of -inf

__device__ __forceinline__ float gelu_f(float x) {
    return 0.5f * x * (1.0f + erff(x * 0.70710678118654752440f));
}
__device__ __forceinline__ unsigned f2u_ord(float f) {
    unsigned u = __float_as_uint(f);
    return (u & 0x80000000u) ? ~u : (u | 0x80000000u);
}
__device__ __forceinline__ float u2f_ord(unsigned e) {
    unsigned b = (e & 0x80000000u) ? (e ^ 0x80000000u) : ~e;
    return __uint_as_float(b);
}

// ---------------------------------------------------------------------------
// Weight packing: W (Cin x Cout) f32 row-major -> f16 WMMA B-fragment layout.
// B fragment (32x16 per k-chunk/col-tile): lane = (k>=16?16:0) + n,
// half-slot h = k%16 (linear-K per lane group, same k order as load_a_frag).
// Packed index: (((kc*ntiles + nt)*32 + lane)*16 + h)
// ---------------------------------------------------------------------------
__global__ void pack_w_kernel(const float* __restrict__ W, _Float16* __restrict__ Wp,
                              int Cin, int Cout) {
    int Kpad = (Cin + 31) & ~31;
    int ntl = Cout >> 4;
    int tot = (Kpad >> 5) * ntl * 32;
    int t = blockIdx.x * blockDim.x + threadIdx.x;
    if (t >= tot) return;
    int lane = t & 31;
    int q = t >> 5;
    int nt = q % ntl;
    int kc = q / ntl;
    int n = nt * 16 + (lane & 15);
    int kbase = kc * 32 + ((lane >= 16) ? 16 : 0);
#pragma unroll
    for (int h = 0; h < 16; ++h) {
        int k = kbase + h;
        float v = (k < Cin) ? W[(size_t)k * Cout + n] : 0.0f;
        Wp[((size_t)t << 4) + h] = (_Float16)v;
    }
}

// ---------------------------------------------------------------------------
// A fragment load from LDS row-major [rows x Kpad] f16 tile.
// A layout (16x32, contiguous k per lane half, same as packed B): lane m = l&15, hi = l>>4;
// a[0..7]  = A[m, kc*32 + hi*16 + 0..7]
// a[8..15] = A[m, kc*32 + hi*16 + 8 + 0..7]
// ---------------------------------------------------------------------------
__device__ __forceinline__ v16h load_a_frag(const _Float16* As, int Kpad,
                                            int m, int hi, int kc) {
    const _Float16* p = As + (size_t)m * Kpad + kc * 32 + hi * 16;
    v8h lo = *(const v8h*)(p);
    v8h hv = *(const v8h*)(p + 8);
    v16h a;
#pragma unroll
    for (int q = 0; q < 8; ++q) { a[q] = lo[q]; a[q + 8] = hv[q]; }
    return a;
}

// ===========================================================================
// Generic fused dense MLP: Y = act(LN(X @ W + b) * g + be)
// 32 rows per block (2 row-tiles share each B fragment), 128 threads.
// ===========================================================================
__global__ __launch_bounds__(128) void dense_mlp_kernel(
    const float* __restrict__ X, int rows, int Cin, int Cout,
    const _Float16* __restrict__ Wp, const float* __restrict__ bias,
    const float* __restrict__ gg, const float* __restrict__ bb,
    int act, float* __restrict__ Y) {
    const int Kpad = (Cin + 31) & ~31;
    __shared__ __align__(16) _Float16 As[32 * 512];
    __shared__ float Of[32 * 512];
    __shared__ float s_mean[32], s_rstd[32];

    const int tid = threadIdx.x;
    const int lane = tid & 31, wv = tid >> 5;
    const int n16 = lane & 15, hi = (lane >> 4) & 1;
    const int r0b = blockIdx.x * 32;

    for (int e = tid; e < 32 * Kpad; e += 128) {
        int row = e / Kpad, kk = e - row * Kpad;
        int r0 = r0b + row;
        float v = (r0 < rows && kk < Cin) ? X[(size_t)r0 * Cin + kk] : 0.0f;
        As[e] = (_Float16)v;
    }
    __syncthreads();

    const int ntl = Cout >> 4, kch = Kpad >> 5;
    for (int nt = wv; nt < ntl; nt += 4) {
        v8f acc0 = {}, acc1 = {};
        for (int kc = 0; kc < kch; ++kc) {
            v16h bf = *(const v16h*)(Wp + ((size_t)(kc * ntl + nt) * 32 + lane) * 16);
            v16h a0 = load_a_frag(As, Kpad, n16, hi, kc);
            v16h a1 = load_a_frag(As, Kpad, n16 + 16, hi, kc);
            acc0 = __builtin_amdgcn_wmma_f32_16x16x32_f16(false, a0, false, bf,
                                                          (short)0, acc0, false, false);
            acc1 = __builtin_amdgcn_wmma_f32_16x16x32_f16(false, a1, false, bf,
                                                          (short)0, acc1, false, false);
        }
        float bv = bias[nt * 16 + n16];
#pragma unroll
        for (int v = 0; v < 8; ++v) {
            Of[(v + hi * 8) * Cout + nt * 16 + n16] = acc0[v] + bv;
            Of[(16 + v + hi * 8) * Cout + nt * 16 + n16] = acc1[v] + bv;
        }
    }
    __syncthreads();

    if (tid < 32) {
        float s = 0.f, ss = 0.f;
        for (int c = 0; c < Cout; ++c) { float x = Of[tid * Cout + c]; s += x; ss += x * x; }
        float mean = s / (float)Cout;
        float var = ss / (float)Cout - mean * mean;
        s_mean[tid] = mean;
        s_rstd[tid] = rsqrtf(var + 1e-5f);
    }
    __syncthreads();

    for (int e = tid; e < 32 * Cout; e += 128) {
        int row = e / Cout, col = e - row * Cout;
        int r0 = r0b + row;
        if (r0 >= rows) continue;
        float x = (Of[e] - s_mean[row]) * s_rstd[row] * gg[col] + bb[col];
        if (act) x = gelu_f(x);
        Y[(size_t)r0 * Cout + col] = x;
    }
}

// ===========================================================================
// Fused local aggregation:
//   rows = B*M*K; row r -> (b, center c, neighbor j); i = idx[b,c,j]
//   A row = [rel(3), feat[b,i,:C], pe(16)]  (pe = rel @ peW + peb, scalar)
//   m0: WMMA -> LN -> GELU ;  m1: WMMA -> LN ;  atomic-max into nf_u[b,c,:]
// 32 rows per block; each B fragment reused by two row-tiles.
// ===========================================================================
__global__ __launch_bounds__(128) void local_agg_kernel(
    const float* __restrict__ pos, const float* __restrict__ centers,
    const float* __restrict__ feat, const int* __restrict__ idx,
    const float* __restrict__ peW, const float* __restrict__ peb,
    const _Float16* __restrict__ Wp0, const float* __restrict__ b0,
    const float* __restrict__ g0, const float* __restrict__ be0,
    const _Float16* __restrict__ Wp1, const float* __restrict__ b1,
    const float* __restrict__ g1, const float* __restrict__ be1,
    unsigned* __restrict__ nf_u,
    int Nn, int Mm, int Kk, int Cc, int C1, int C2) {
    const int Cin0 = 3 + Cc + 16;
    const int Kpad0 = (Cin0 + 31) & ~31;
    __shared__ __align__(16) _Float16 A0[32 * 288];
    __shared__ float Of[32 * 512];           // reused: m0 out (C1), then m1 out (C2)
    __shared__ __align__(16) _Float16 A1[32 * 512];
    __shared__ float s_rel[32][3];
    __shared__ int s_nb[32];
    __shared__ int s_ctr[32];
    __shared__ float s_mean[32], s_rstd[32];

    const int tid = threadIdx.x;
    const int lane = tid & 31, wv = tid >> 5;
    const int n16 = lane & 15, hi = (lane >> 4) & 1;

    if (tid < 32) {
        int r0 = blockIdx.x * 32 + tid;
        int c = r0 / Kk, j = r0 - c * Kk;
        int b = c / Mm, cc = c - b * Mm;
        int i = idx[((size_t)b * Mm + cc) * Kk + j];
        s_nb[tid] = b * Nn + i;
        s_ctr[tid] = b * Mm + cc;
#pragma unroll
        for (int d = 0; d < 3; ++d)
            s_rel[tid][d] = pos[((size_t)b * Nn + i) * 3 + d] -
                            centers[((size_t)b * Mm + cc) * 3 + d];
    }
    __syncthreads();

    // stage gathered A0 (32 x Kpad0) f16
    for (int e = tid; e < 32 * Kpad0; e += 128) {
        int row = e / Kpad0, kk = e - row * Kpad0;
        float v = 0.0f;
        if (kk < 3) v = s_rel[row][kk];
        else if (kk < 3 + Cc) v = feat[(size_t)s_nb[row] * Cc + (kk - 3)];
        else if (kk < Cin0) {
            int q = kk - 3 - Cc;
            v = peb[q] + s_rel[row][0] * peW[q] + s_rel[row][1] * peW[16 + q] +
                s_rel[row][2] * peW[32 + q];
        }
        A0[e] = (_Float16)v;
    }
    __syncthreads();

    // ---- m0 ----
    const int ntl0 = C1 >> 4, kch0 = Kpad0 >> 5;
    for (int nt = wv; nt < ntl0; nt += 4) {
        v8f acc0 = {}, acc1 = {};
        for (int kc = 0; kc < kch0; ++kc) {
            v16h bf = *(const v16h*)(Wp0 + ((size_t)(kc * ntl0 + nt) * 32 + lane) * 16);
            v16h a0 = load_a_frag(A0, Kpad0, n16, hi, kc);
            v16h a1 = load_a_frag(A0, Kpad0, n16 + 16, hi, kc);
            acc0 = __builtin_amdgcn_wmma_f32_16x16x32_f16(false, a0, false, bf,
                                                          (short)0, acc0, false, false);
            acc1 = __builtin_amdgcn_wmma_f32_16x16x32_f16(false, a1, false, bf,
                                                          (short)0, acc1, false, false);
        }
        float bv = b0[nt * 16 + n16];
#pragma unroll
        for (int v = 0; v < 8; ++v) {
            Of[(v + hi * 8) * C1 + nt * 16 + n16] = acc0[v] + bv;
            Of[(16 + v + hi * 8) * C1 + nt * 16 + n16] = acc1[v] + bv;
        }
    }
    __syncthreads();

    if (tid < 32) {
        float s = 0.f, ss = 0.f;
        for (int c = 0; c < C1; ++c) { float x = Of[tid * C1 + c]; s += x; ss += x * x; }
        float mean = s / (float)C1;
        float var = ss / (float)C1 - mean * mean;
        s_mean[tid] = mean;
        s_rstd[tid] = rsqrtf(var + 1e-5f);
    }
    __syncthreads();

    for (int e = tid; e < 32 * C1; e += 128) {
        int row = e / C1, col = e - row * C1;
        float x = (Of[e] - s_mean[row]) * s_rstd[row] * g0[col] + be0[col];
        A1[e] = (_Float16)gelu_f(x);
    }
    __syncthreads();

    // ---- m1 ---- (Of reused with row-stride C2)
    const int ntl1 = C2 >> 4, kch1 = C1 >> 5;
    for (int nt = wv; nt < ntl1; nt += 4) {
        v8f acc0 = {}, acc1 = {};
        for (int kc = 0; kc < kch1; ++kc) {
            v16h bf = *(const v16h*)(Wp1 + ((size_t)(kc * ntl1 + nt) * 32 + lane) * 16);
            v16h a0 = load_a_frag(A1, C1, n16, hi, kc);
            v16h a1 = load_a_frag(A1, C1, n16 + 16, hi, kc);
            acc0 = __builtin_amdgcn_wmma_f32_16x16x32_f16(false, a0, false, bf,
                                                          (short)0, acc0, false, false);
            acc1 = __builtin_amdgcn_wmma_f32_16x16x32_f16(false, a1, false, bf,
                                                          (short)0, acc1, false, false);
        }
        float bv = b1[nt * 16 + n16];
#pragma unroll
        for (int v = 0; v < 8; ++v) {
            Of[(v + hi * 8) * C2 + nt * 16 + n16] = acc0[v] + bv;
            Of[(16 + v + hi * 8) * C2 + nt * 16 + n16] = acc1[v] + bv;
        }
    }
    __syncthreads();

    if (tid < 32) {
        float s = 0.f, ss = 0.f;
        for (int c = 0; c < C2; ++c) { float x = Of[tid * C2 + c]; s += x; ss += x * x; }
        float mean = s / (float)C2;
        float var = ss / (float)C2 - mean * mean;
        s_mean[tid] = mean;
        s_rstd[tid] = rsqrtf(var + 1e-5f);
    }
    __syncthreads();

    for (int e = tid; e < 32 * C2; e += 128) {
        int row = e / C2, col = e - row * C2;
        float x = (Of[e] - s_mean[row]) * s_rstd[row] * g1[col] + be1[col];
        atomicMax(&nf_u[(size_t)s_ctr[row] * C2 + col], f2u_ord(x));
    }
}

// ===========================================================================
// Farthest point sampling: one block per batch, sequential argmax loop.
// ===========================================================================
__global__ __launch_bounds__(256) void fps_kernel(const float* __restrict__ pos,
                                                  int Nn, int Mn, int* __restrict__ sidx) {
    __shared__ float dmin[4096];
    __shared__ float rv[256];
    __shared__ int ri[256];
    __shared__ int s_last;
    const int b = blockIdx.x, tid = threadIdx.x;
    for (int i = tid; i < Nn; i += 256) dmin[i] = 1e10f;
    if (tid == 0) { sidx[b * Mn] = 0; s_last = 0; }
    __syncthreads();
    int last = 0;
    for (int s = 1; s < Mn; ++s) {
        const float lx = pos[((size_t)b * Nn + last) * 3 + 0];
        const float ly = pos[((size_t)b * Nn + last) * 3 + 1];
        const float lz = pos[((size_t)b * Nn + last) * 3 + 2];
        float bestv = -1.0f; int besti = 0;
        for (int i = tid; i < Nn; i += 256) {
            float dx = pos[((size_t)b * Nn + i) * 3 + 0] - lx;
            float dy = pos[((size_t)b * Nn + i) * 3 + 1] - ly;
            float dz = pos[((size_t)b * Nn + i) * 3 + 2] - lz;
            float d = dx * dx + dy * dy + dz * dz;
            float dm = fminf(dmin[i], d);
            dmin[i] = dm;
            if (dm > bestv) { bestv = dm; besti = i; }
        }
        rv[tid] = bestv; ri[tid] = besti;
        __syncthreads();
        for (int sft = 128; sft > 0; sft >>= 1) {
            if (tid < sft) {
                float ov = rv[tid + sft]; int oi = ri[tid + sft];
                if (ov > rv[tid] || (ov == rv[tid] && oi < ri[tid])) { rv[tid] = ov; ri[tid] = oi; }
            }
            __syncthreads();
        }
        if (tid == 0) { sidx[b * Mn + s] = ri[0]; s_last = ri[0]; }
        __syncthreads();
        last = s_last;
    }
}

// ===========================================================================
// Ball query: first k indices (ascending) with d2 <= r2, padded with first.
// ===========================================================================
__global__ void ball_query_kernel(const float* __restrict__ centers,
                                  const float* __restrict__ pos,
                                  int Bn, int Mn, int Nn, int Kk, float r2,
                                  int* __restrict__ idx) {
    int t = blockIdx.x * blockDim.x + threadIdx.x;
    if (t >= Bn * Mn) return;
    int b = t / Mn;
    float cx = centers[(size_t)t * 3 + 0];
    float cy = centers[(size_t)t * 3 + 1];
    float cz = centers[(size_t)t * 3 + 2];
    const float* pb = pos + (size_t)b * Nn * 3;
    int cnt = 0, first = 0; bool have = false;
    for (int i = 0; i < Nn && cnt < Kk; ++i) {
        float dx = pb[i * 3 + 0] - cx;
        float dy = pb[i * 3 + 1] - cy;
        float dz = pb[i * 3 + 2] - cz;
        float d2 = dx * dx + dy * dy + dz * dz;
        if (d2 <= r2) {
            if (!have) { first = i; have = true; }
            idx[(size_t)t * Kk + cnt] = i;
            ++cnt;
        }
    }
    for (int q = cnt; q < Kk; ++q) idx[(size_t)t * Kk + q] = first;
}

// ----------------------------- small kernels -------------------------------
__global__ void extract_pos_kernel(const float* __restrict__ pc, float* __restrict__ pos, int tot) {
    int t = blockIdx.x * blockDim.x + threadIdx.x;
    if (t >= tot) return;
    pos[(size_t)t * 3 + 0] = pc[(size_t)t * 6 + 0];
    pos[(size_t)t * 3 + 1] = pc[(size_t)t * 6 + 1];
    pos[(size_t)t * 3 + 2] = pc[(size_t)t * 6 + 2];
}
__global__ void gather_rows_kernel(const float* __restrict__ src, const int* __restrict__ sidx,
                                   float* __restrict__ dst, int Bn, int Nn, int Mn, int Cn) {
    size_t t = (size_t)blockIdx.x * blockDim.x + threadIdx.x;
    size_t tot = (size_t)Bn * Mn * Cn;
    if (t >= tot) return;
    int c = (int)(t % Cn);
    size_t bm = t / Cn;
    int m = (int)(bm % Mn);
    int b = (int)(bm / Mn);
    int i = sidx[b * Mn + m];
    dst[t] = src[((size_t)b * Nn + i) * Cn + c];
}
__global__ void fill_u32_kernel(unsigned* __restrict__ p, unsigned v, int n) {
    int t = blockIdx.x * blockDim.x + threadIdx.x;
    if (t < n) p[t] = v;
}
__global__ void u2f_kernel(const unsigned* __restrict__ a, float* __restrict__ o, int n) {
    int t = blockIdx.x * blockDim.x + threadIdx.x;
    if (t < n) o[t] = u2f_ord(a[t]);
}
__global__ void add_gelu_u_kernel(const unsigned* __restrict__ a, const float* __restrict__ b,
                                  float* __restrict__ o, int n) {
    int t = blockIdx.x * blockDim.x + threadIdx.x;
    if (t < n) o[t] = gelu_f(u2f_ord(a[t]) + b[t]);
}
__global__ void add_gelu_f_kernel(const float* __restrict__ a, const float* __restrict__ b,
                                  float* __restrict__ o, int n) {
    int t = blockIdx.x * blockDim.x + threadIdx.x;
    if (t < n) o[t] = gelu_f(a[t] + b[t]);
}
__global__ void add_kernel(const float* __restrict__ a, const float* __restrict__ b,
                           float* __restrict__ o, int n) {
    int t = blockIdx.x * blockDim.x + threadIdx.x;
    if (t < n) o[t] = a[t] + b[t];
}
__global__ void reduce_max_kernel(const float* __restrict__ f, float* __restrict__ o,
                                  int Bn, int Mn, int Cn) {
    int t = blockIdx.x * blockDim.x + threadIdx.x;
    if (t >= Bn * Cn) return;
    int b = t / Cn, ch = t - b * Cn;
    float m = -INFINITY;
    for (int i = 0; i < Mn; ++i) m = fmaxf(m, f[((size_t)b * Mn + i) * Cn + ch]);
    o[t] = m;
}
__global__ void mean_pos_kernel(const float* __restrict__ pos, float* __restrict__ ppos,
                                int Bn, int Mn) {
    int t = blockIdx.x * blockDim.x + threadIdx.x;
    if (t >= Bn * 3) return;
    int b = t / 3, d = t - b * 3;
    float s = 0.f;
    for (int i = 0; i < Mn; ++i) s += pos[((size_t)b * Mn + i) * 3 + d];
    ppos[t] = s / (float)Mn;
}
__global__ void sin_emb_kernel(const float* __restrict__ ppos, float* __restrict__ semb) {
    int t = blockIdx.x * blockDim.x + threadIdx.x;
    if (t >= 8 * 48) return;
    int b = t / 48, r = t - b * 48, d = r / 16, q = r - d * 16;
    float f = expf(-logf(10000.0f) * (float)q / 15.0f);
    float a = ppos[b * 3 + d] * f;
    semb[b * 96 + d * 32 + q] = sinf(a);
    semb[b * 96 + d * 32 + 16 + q] = cosf(a);
}

// ===========================================================================
// Host orchestration
// ===========================================================================
extern "C" void kernel_launch(void* const* d_in, const int* in_sizes, int n_in,
                              void* d_out, int out_size, void* d_ws, size_t ws_size,
                              hipStream_t stream) {
    (void)in_sizes; (void)n_in; (void)out_size; (void)ws_size;
    const int B = 8, N1 = 4096, N2 = 2048, N3 = 1024;
    const float* pc = (const float*)d_in[0];
    auto F = [&](int i) { return (const float*)d_in[i]; };

    char* base = (char*)d_ws;
    size_t off = 0;
    auto alloc = [&](size_t bytes) -> void* {
        off = (off + 255) & ~(size_t)255;
        void* p = base + off;
        off += bytes;
        return p;
    };

    struct Lay { const float *W, *b, *g, *be; int cin, cout; _Float16* wp; };
    auto mk = [&](int wi, int cin, int cout) -> Lay {
        Lay l;
        l.W = F(wi); l.b = F(wi + 1); l.g = F(wi + 2); l.be = F(wi + 3);
        l.cin = cin; l.cout = cout;
        int kpad = (cin + 31) & ~31;
        l.wp = (_Float16*)alloc((size_t)kpad * cout * 2);
        int tot = (kpad >> 5) * (cout >> 4) * 32;
        pack_w_kernel<<<(tot + 255) / 256, 256, 0, stream>>>(l.W, l.wp, cin, cout);
        return l;
    };

    // input index map (insertion order of setup_inputs dicts)
    Lay s0   = mk(1, 6, 64);
    const float *pe1W = F(5),  *pe1b = F(6);
    Lay s1m0 = mk(7, 83, 64),  s1m1 = mk(11, 64, 128),  s1res = mk(15, 64, 128);
    const float *pe1bW = F(19), *pe1bb = F(20);
    Lay b1m0 = mk(21, 147, 256), b1m1 = mk(25, 256, 128);
    Lay b1c0 = mk(29, 128, 256), b1c1 = mk(33, 256, 128);
    const float *pe2W = F(37), *pe2b = F(38);
    Lay s2m0 = mk(39, 147, 128), s2m1 = mk(43, 128, 256), s2res = mk(47, 128, 256);
    const float *pe2bW = F(51), *pe2bb = F(52);
    Lay b2m0 = mk(53, 275, 512), b2m1 = mk(57, 512, 256);
    Lay b2c0 = mk(61, 256, 512), b2c1 = mk(65, 512, 256);
    Lay gp   = mk(69, 96, 256),  outl = mk(73, 256, 256);

    float* pos1 = (float*)alloc((size_t)B * N1 * 3 * 4);
    float* pos2 = (float*)alloc((size_t)B * N2 * 3 * 4);
    float* pos3 = (float*)alloc((size_t)B * N3 * 3 * 4);
    int* sidx   = (int*)alloc((size_t)B * N2 * 4);
    int* idx    = (int*)alloc((size_t)B * N2 * 24 * 4);
    unsigned* nfu = (unsigned*)alloc((size_t)B * N2 * 128 * 4);
    float* featA = (float*)alloc((size_t)B * N1 * 64 * 4);
    float* featB = (float*)alloc((size_t)B * N1 * 64 * 4);
    float* scr1 = (float*)alloc((size_t)2097152 * 4);
    float* scr2 = (float*)alloc((size_t)4194304 * 4);
    float* scr3 = (float*)alloc((size_t)2097152 * 4);
    float* pooled = (float*)alloc((size_t)B * 256 * 4);
    float* ppos   = (float*)alloc((size_t)B * 3 * 4);
    float* semb   = (float*)alloc((size_t)B * 96 * 4);
    float* gout   = (float*)alloc((size_t)B * 256 * 4);
    float* gbuf   = (float*)alloc((size_t)B * 256 * 4);

    auto dense = [&](const float* X, int rows, const Lay& l, int act, float* Y) {
        dense_mlp_kernel<<<(rows + 31) / 32, 128, 0, stream>>>(
            X, rows, l.cin, l.cout, l.wp, l.b, l.g, l.be, act, Y);
    };
    auto agg = [&](const float* pos, const float* ctr, const float* feat, const int* id,
                   const float* peW, const float* peb, const Lay& m0, const Lay& m1,
                   unsigned* nf, int Nn, int Mm, int Kk, int Cc) {
        int rows = B * Mm * Kk;
        local_agg_kernel<<<rows / 32, 128, 0, stream>>>(
            pos, ctr, feat, id, peW, peb,
            m0.wp, m0.b, m0.g, m0.be, m1.wp, m1.b, m1.g, m1.be,
            nf, Nn, Mm, Kk, Cc, m0.cout, m1.cout);
    };
    auto fill = [&](unsigned* p, int n) {
        fill_u32_kernel<<<(n + 255) / 256, 256, 0, stream>>>(p, NEG_ENC, n);
    };

    // ---- stem ----
    extract_pos_kernel<<<(B * N1 + 255) / 256, 256, 0, stream>>>(pc, pos1, B * N1);
    dense(pc, B * N1, s0, 1, featA);

    // ---- stage 1: set_abs (stride 2, r=0.08, k=24) ----
    fps_kernel<<<B, 256, 0, stream>>>(pos1, N1, N2, sidx);
    gather_rows_kernel<<<(B * N2 * 3 + 255) / 256, 256, 0, stream>>>(pos1, sidx, pos2, B, N1, N2, 3);
    gather_rows_kernel<<<(B * N2 * 64 + 255) / 256, 256, 0, stream>>>(featA, sidx, scr1, B, N1, N2, 64);
    ball_query_kernel<<<(B * N2 + 127) / 128, 128, 0, stream>>>(pos2, pos1, B, N2, N1, 24, 0.0064f, idx);
    fill(nfu, B * N2 * 128);
    agg(pos1, pos2, featA, idx, pe1W, pe1b, s1m0, s1m1, nfu, N1, N2, 24, 64);
    dense(scr1, B * N2, s1res, 0, scr2);
    add_gelu_u_kernel<<<(B * N2 * 128 + 255) / 256, 256, 0, stream>>>(nfu, scr2, featB, B * N2 * 128);

    // ---- stage 1: inv_res ----
    ball_query_kernel<<<(B * N2 + 127) / 128, 128, 0, stream>>>(pos2, pos2, B, N2, N2, 24, 0.0064f, idx);
    fill(nfu, B * N2 * 128);
    agg(pos2, pos2, featB, idx, pe1bW, pe1bb, b1m0, b1m1, nfu, N2, N2, 24, 128);
    u2f_kernel<<<(B * N2 * 128 + 255) / 256, 256, 0, stream>>>(nfu, scr1, B * N2 * 128);
    dense(scr1, B * N2, b1c0, 1, scr2);
    dense(scr2, B * N2, b1c1, 0, scr3);
    add_gelu_f_kernel<<<(B * N2 * 128 + 255) / 256, 256, 0, stream>>>(scr3, featB, featA, B * N2 * 128);

    // ---- stage 2: set_abs (stride 2, r=0.16, k=32) ----
    fps_kernel<<<B, 256, 0, stream>>>(pos2, N2, N3, sidx);
    gather_rows_kernel<<<(B * N3 * 3 + 255) / 256, 256, 0, stream>>>(pos2, sidx, pos3, B, N2, N3, 3);
    gather_rows_kernel<<<(B * N3 * 128 + 255) / 256, 256, 0, stream>>>(featA, sidx, scr1, B, N2, N3, 128);
    ball_query_kernel<<<(B * N3 + 127) / 128, 128, 0, stream>>>(pos3, pos2, B, N3, N2, 32, 0.0256f, idx);
    fill(nfu, B * N3 * 256);
    agg(pos2, pos3, featA, idx, pe2W, pe2b, s2m0, s2m1, nfu, N2, N3, 32, 128);
    dense(scr1, B * N3, s2res, 0, scr2);
    add_gelu_u_kernel<<<(B * N3 * 256 + 255) / 256, 256, 0, stream>>>(nfu, scr2, featB, B * N3 * 256);

    // ---- stage 2: inv_res ----
    ball_query_kernel<<<(B * N3 + 127) / 128, 128, 0, stream>>>(pos3, pos3, B, N3, N3, 32, 0.0256f, idx);
    fill(nfu, B * N3 * 256);
    agg(pos3, pos3, featB, idx, pe2bW, pe2bb, b2m0, b2m1, nfu, N3, N3, 32, 256);
    u2f_kernel<<<(B * N3 * 256 + 255) / 256, 256, 0, stream>>>(nfu, scr1, B * N3 * 256);
    dense(scr1, B * N3, b2c0, 1, scr2);
    dense(scr2, B * N3, b2c1, 0, scr3);
    add_gelu_f_kernel<<<(B * N3 * 256 + 255) / 256, 256, 0, stream>>>(scr3, featB, featA, B * N3 * 256);

    // ---- head ----
    reduce_max_kernel<<<(B * 256 + 255) / 256, 256, 0, stream>>>(featA, pooled, B, N3, 256);
    mean_pos_kernel<<<1, 32, 0, stream>>>(pos3, ppos, B, N3);
    sin_emb_kernel<<<2, 256, 0, stream>>>(ppos, semb);
    dense(semb, B, gp, 1, gout);
    add_kernel<<<(B * 256 + 255) / 256, 256, 0, stream>>>(pooled, gout, gbuf, B * 256);
    dense(gbuf, B, outl, 0, (float*)d_out);
}
